// MRConv4dSP_35493609734669
// MI455X (gfx1250) — matrix operand order repaired and checked by reading, not compiled, not run on hardware
//
#include <hip/hip_runtime.h>
#include <math.h>

typedef __attribute__((ext_vector_type(16))) _Float16 v16h;
typedef __attribute__((ext_vector_type(8)))  _Float16 v8h;
typedef __attribute__((ext_vector_type(8)))  float    v8f;

#define B_   8
#define C_   96
#define H_   224
#define W_   224
#define HW   (H_*W_)      // 50176
#define SH   56
#define N_   (SH*SH)      // 3136
#define OUT_ 192
#define NT   (N_/16)      // 196 tiles per batch
#define LOG2E 1.4426950408889634f

// ---------------- workspace layout (bytes) ----------------
// q16   : (B, N, C)   f16  4,816,896
// qT16  : (B, C, N)   f16  4,816,896
// att16 : (B, N, C)   f16  4,816,896
// xT16  : (B, HW, C)  f16  77,070,336
// w16   : (OUT, 2C)   f16  73,728
#define OFF_QT   4816896u
#define OFF_AT   9633792u
#define OFF_XT   14450688u
#define OFF_W16  91521024u

static __device__ __forceinline__ v16h cat8(v8h lo, v8h hi) {
  return __builtin_shufflevector(lo, hi, 0,1,2,3,4,5,6,7,8,9,10,11,12,13,14,15);
}

static __device__ __forceinline__ void wave_lds_fence() {
  __threadfence_block();              // compiler/memory ordering for LDS
  __builtin_amdgcn_wave_barrier();    // scheduling barrier, no cross-wave sync
}

// ---------------- kernel 1: strided extract -> q (f16) + qT (f16) ----------
__global__ void k_extract(const float* __restrict__ x,
                          _Float16* __restrict__ q16,
                          _Float16* __restrict__ qT16) {
  int idx = blockIdx.x * 256 + threadIdx.x;
  if (idx >= B_ * C_ * N_) return;
  int n = idx % N_;
  int t = idx / N_;
  int c = t % C_;
  int b = t / C_;
  int i = n / SH, j = n % SH;
  float v = x[(((size_t)b * C_ + c) * H_ + i * 4) * W_ + j * 4];
  _Float16 h = (_Float16)v;
  q16 [((size_t)b * N_ + n) * C_ + c] = h;
  qT16[((size_t)b * C_ + c) * N_ + n] = h;
}

// ---------------- kernel 1b: conv_w fp32 -> f16 ----------------------------
__global__ void k_cvtw(const float* __restrict__ w, _Float16* __restrict__ w16) {
  int idx = blockIdx.x * 256 + threadIdx.x;
  if (idx < OUT_ * 2 * C_) w16[idx] = (_Float16)w[idx];
}

// ---------------- kernel 1c: x (b,c,p) f32 -> xT16 (b,p,c) f16 -------------
// one block: 96 channels x 64 pixels, LDS tiled, coalesced both directions
__launch_bounds__(256)
__global__ void k_xpose(const float* __restrict__ x, _Float16* __restrict__ xT16) {
  __shared__ float tile[C_][65];                 // 65: conflict-free transpose
  const int b  = blockIdx.x / (HW / 64);
  const int p0 = (blockIdx.x % (HW / 64)) * 64;
  const float* xb = x + (size_t)b * C_ * HW + p0;
#pragma unroll
  for (int it = 0; it < (C_ * 64) / 256; ++it) { // 24 iters
    int lin = it * 256 + threadIdx.x;
    int c = lin >> 6, dp = lin & 63;
    tile[c][dp] = xb[(size_t)c * HW + dp];
  }
  __syncthreads();
  _Float16* ob = xT16 + ((size_t)b * HW + p0) * C_;
#pragma unroll
  for (int it = 0; it < (C_ * 64) / 256; ++it) { // 24 iters
    int lin = it * 256 + threadIdx.x;
    int dp = lin / C_, c = lin % C_;
    ob[(size_t)dp * C_ + c] = (_Float16)tile[c][dp];
  }
}

// ---------------- kernel 2: two-sweep flash attention ----------------------
__launch_bounds__(256)
__global__ void k_attn(const _Float16* __restrict__ q16,
                       const _Float16* __restrict__ qT16,
                       _Float16* __restrict__ att16) {
  __shared__ __align__(32) _Float16 pbuf[8][16][32];   // per-wave P tile

  const int lane = threadIdx.x & 31;
  const int wv   = threadIdx.x >> 5;
  const int half = lane >> 4;
  const int lr   = lane & 15;
  const int t    = blockIdx.x * 8 + wv;     // 0..1567
  const int b    = t / NT;
  const int m0   = (t % NT) * 16;
  const float kexp = rsqrtf((float)C_) * LOG2E;   // fold scale + log2(e)

  const _Float16* qb  = q16  + (size_t)b * N_ * C_;
  const _Float16* qTb = qT16 + (size_t)b * C_ * N_;

  // resident A fragments: Q rows m0..m0+15
  const _Float16* arow = qb + (size_t)(m0 + lr) * C_;
  v16h aq[3];
#pragma unroll
  for (int kc = 0; kc < 3; ++kc) {
    aq[kc] = *(const v16h*)(arow + kc * 32 + half * 16);
  }

  // ---- sweep 1: per-lane row maxima (no cross-lane work in the loop) ----
  v8f mx;
#pragma unroll
  for (int r = 0; r < 8; ++r) mx[r] = -INFINITY;
  for (int nt = 0; nt < NT; ++nt) {
    const _Float16* brow = qb + (size_t)(nt * 16 + lr) * C_;
    v8f s = {};
#pragma unroll
    for (int kc = 0; kc < 3; ++kc) {
      v16h bq = *(const v16h*)(brow + kc * 32 + half * 16);
      s = __builtin_amdgcn_wmma_f32_16x16x32_f16(false, aq[kc], false, bq,
                                                 (short)0, s, false, false);
    }
#pragma unroll
    for (int r = 0; r < 8; ++r) mx[r] = fmaxf(mx[r], s[r]);
  }
  // single cross-lane max reduction (16-lane halves), then pre-scale
  v8f mb;
#pragma unroll
  for (int r = 0; r < 8; ++r) {
    float v = mx[r];
    v = fmaxf(v, __shfl_xor(v, 1, 32));
    v = fmaxf(v, __shfl_xor(v, 2, 32));
    v = fmaxf(v, __shfl_xor(v, 4, 32));
    v = fmaxf(v, __shfl_xor(v, 8, 32));
    mb[r] = v * kexp;                 // exp2 argument offset
  }

  // ---- sweep 2: P = exp2(s*k - mb); l += rowsum(P); O += P*V ----
  v8f oacc[6] = {};
  v8f lsum = {};
  for (int ch = 0; ch < NT / 2; ++ch) {         // 98 chunks of 32 columns
    const int n0 = ch * 32;
#pragma unroll
    for (int t2 = 0; t2 < 2; ++t2) {
      const _Float16* brow = qb + (size_t)(n0 + t2 * 16 + lr) * C_;
      v8f s = {};
#pragma unroll
      for (int kc = 0; kc < 3; ++kc) {
        v16h bq = *(const v16h*)(brow + kc * 32 + half * 16);
        s = __builtin_amdgcn_wmma_f32_16x16x32_f16(false, aq[kc], false, bq,
                                                   (short)0, s, false, false);
      }
#pragma unroll
      for (int r = 0; r < 8; ++r) {
        float p = exp2f(fmaf(s[r], kexp, -mb[r]));
        lsum[r] += p;
        pbuf[wv][r + 8 * half][t2 * 16 + lr] = (_Float16)p;   // C-layout -> LDS
      }
    }
    wave_lds_fence();
    v16h pa = *(const v16h*)(&pbuf[wv][lr][half * 16]);
    wave_lds_fence();
#pragma unroll
    for (int ct = 0; ct < 6; ++ct) {
      const _Float16* vrow = qTb + (size_t)(ct * 16 + lr) * N_;   // V^T rows
      v16h bv = *(const v16h*)(vrow + n0 + half * 16);
      oacc[ct] = __builtin_amdgcn_wmma_f32_16x16x32_f16(false, pa, false, bv,
                                                        (short)0, oacc[ct],
                                                        false, false);
    }
  }

  // finalize row sums, invert once
#pragma unroll
  for (int r = 0; r < 8; ++r) {
    float v = lsum[r];
    v += __shfl_xor(v, 1, 32);
    v += __shfl_xor(v, 2, 32);
    v += __shfl_xor(v, 4, 32);
    v += __shfl_xor(v, 8, 32);
    lsum[r] = 1.0f / v;
  }

  _Float16* ob = att16 + ((size_t)b * N_ + m0) * C_;
#pragma unroll
  for (int ct = 0; ct < 6; ++ct)
#pragma unroll
    for (int r = 0; r < 8; ++r)
      ob[(size_t)(r + 8 * half) * C_ + ct * 16 + lr] =
          (_Float16)(oacc[ct][r] * lsum[r]);
}

// ---------------- kernel 3: fused 1x1 conv + BN + GELU via WMMA ------------
// one wave = 16 pixels x all 192 outputs; every operand is a contiguous load
__launch_bounds__(256)
__global__ void k_conv(const _Float16* __restrict__ xT16,
                       const _Float16* __restrict__ att16,
                       const _Float16* __restrict__ w16,
                       const float* __restrict__ conv_b,
                       const float* __restrict__ gamma,
                       const float* __restrict__ beta,
                       const float* __restrict__ mean,
                       const float* __restrict__ var,
                       float* __restrict__ out) {
  __shared__ float sA[OUT_], sB[OUT_];
  if (threadIdx.x < OUT_) {
    int o = threadIdx.x;
    float a = gamma[o] * rsqrtf(var[o] + 1e-5f);
    sA[o] = a;
    sB[o] = (conv_b[o] - mean[o]) * a + beta[o];
  }
  __syncthreads();

  const int lane = threadIdx.x & 31;
  const int wv   = threadIdx.x >> 5;
  const int half = lane >> 4;
  const int lr   = lane & 15;
  const int t    = blockIdx.x * 8 + wv;        // 0..25087
  const int b    = t / (HW / 16);
  const int p0   = (t % (HW / 16)) * 16;
  const int p    = p0 + lr;
  const int hh   = p / W_, ww = p % W_;
  const int n    = (hh >> 2) * SH + (ww >> 2);

  const _Float16* xrow = xT16 + ((size_t)b * HW + p) * C_;
  const _Float16* arow = att16 + ((size_t)b * N_ + n) * C_;

  v8f acc[12] = {};
#pragma unroll
  for (int k = 0; k < 6; ++k) {                // K = 192 -> 6 chunks of 32
    v16h bv = (k < 3)
      ? *(const v16h*)(xrow + k * 32 + half * 16)
      : *(const v16h*)(arow + (k - 3) * 32 + half * 16);
#pragma unroll
    for (int ot = 0; ot < 12; ++ot) {
      const _Float16* wrow = w16 + (size_t)(ot * 16 + lr) * (2 * C_);
      v16h aw = *(const v16h*)(wrow + k * 32 + half * 16);
      acc[ot] = __builtin_amdgcn_wmma_f32_16x16x32_f16(false, aw, false, bv,
                                                       (short)0, acc[ot],
                                                       false, false);
    }
  }

  float* ob = out + (size_t)b * OUT_ * HW + p;
#pragma unroll
  for (int ot = 0; ot < 12; ++ot) {
#pragma unroll
    for (int r = 0; r < 8; ++r) {
      int o = ot * 16 + r + 8 * half;
      float y = acc[ot][r] * sA[o] + sB[o];
      float g = 0.5f * y * (1.0f + erff(y * 0.70710678118654752f));
      ob[(size_t)o * HW] = g;
    }
  }
}

// ---------------- host launcher --------------------------------------------
extern "C" void kernel_launch(void* const* d_in, const int* in_sizes, int n_in,
                              void* d_out, int out_size, void* d_ws, size_t ws_size,
                              hipStream_t stream) {
  const float* x      = (const float*)d_in[0];
  const float* conv_w = (const float*)d_in[1];
  const float* conv_b = (const float*)d_in[2];
  const float* gamma  = (const float*)d_in[3];
  const float* beta   = (const float*)d_in[4];
  const float* mean   = (const float*)d_in[5];
  const float* var    = (const float*)d_in[6];

  char* ws = (char*)d_ws;
  _Float16* q16   = (_Float16*)(ws);
  _Float16* qT16  = (_Float16*)(ws + OFF_QT);
  _Float16* att16 = (_Float16*)(ws + OFF_AT);
  _Float16* xT16  = (_Float16*)(ws + OFF_XT);
  _Float16* w16   = (_Float16*)(ws + OFF_W16);

  k_extract<<<(B_ * C_ * N_) / 256, 256, 0, stream>>>(x, q16, qT16);
  k_cvtw<<<(OUT_ * 2 * C_ + 255) / 256, 256, 0, stream>>>(conv_w, w16);
  k_xpose<<<B_ * (HW / 64), 256, 0, stream>>>(x, xT16);
  k_attn<<<196, 256, 0, stream>>>(q16, qT16, att16);
  k_conv<<<3136, 256, 0, stream>>>(xT16, att16, w16, conv_b, gamma, beta,
                                   mean, var, (float*)d_out);
}
